// ConvLSTM_77060303225570
// MI455X (gfx1250) — compile-verified, matrix operand order checked by reading, not run on hardware
//
#include <hip/hip_runtime.h>
#include <hip/hip_bf16.h>
#include <cstdint>
#include <cstddef>

typedef __attribute__((ext_vector_type(16))) _Float16 v16h;
typedef __attribute__((ext_vector_type(8)))  _Float16 v8h;
typedef __attribute__((ext_vector_type(8)))  float    v8f;
typedef int v4i __attribute__((vector_size(16)));

namespace {
constexpr int kB = 16, kH = 128, kW = 128;
constexpr int kPix = kB * kH * kW;    // 262144
constexpr int kNGate = 320;           // ci,co,cg,px,py  x 64 channels
constexpr int kBS = 40;               // staged B row stride (halfs); 80B -> 16B
                                      // aligned + conflict-free b128 pattern
}

#ifndef __has_builtin
#define __has_builtin(x) 0
#endif
#if __has_builtin(__builtin_amdgcn_global_load_async_to_lds_b128) && \
    __has_builtin(__builtin_amdgcn_s_wait_asynccnt)
#define USE_ASYNC_LDS 1
#else
#define USE_ASYNC_LDS 0
#endif

#define AS1 __attribute__((address_space(1)))
#define AS3 __attribute__((address_space(3)))

// 16-byte global -> LDS copy. Async (ASYNCcnt) when available.
__device__ __forceinline__ void cp16_g2l(const _Float16* __restrict__ g,
                                         _Float16* __restrict__ l) {
#if USE_ASYNC_LDS
  __builtin_amdgcn_global_load_async_to_lds_b128((AS1 v4i*)g, (AS3 v4i*)l, 0,
                                                 0);
#else
  *(v8h*)l = *(const v8h*)g;
#endif
}
__device__ __forceinline__ void cp_g2l_wait() {
#if USE_ASYNC_LDS
  __builtin_amdgcn_s_wait_asynccnt(0);
#endif
}

__device__ __forceinline__ v8f wmma32(v16h a, v16h b, v8f c) {
  // D = A(16x32 f16) x B(32x16 f16) + C(16x16 f32)
  return __builtin_amdgcn_wmma_f32_16x16x32_f16(false, a, false, b, (short)0, c,
                                                false, false);
}

__device__ __forceinline__ float sigmoidf(float x) {
  return 1.0f / (1.0f + __expf(-x));
}

// ---- fragment helpers (ISA 7.12.2 layouts, wave32) --------------------------
// A 16x32 (or B as W^T with lane = output column): lane r=lane&15, hh=lane>>4,
// VGPR v<4 -> K = 8*hh + 2v, v>=4 -> K = 16 + 8*hh + 2(v-4). Pairs contiguous.
__device__ __forceinline__ v16h frag_ld_rm(const _Float16* __restrict__ src,
                                           int ld, int rbase, int cbase) {
  const int lane = threadIdx.x & 31;
  const int r = lane & 15, hh = lane >> 4;
  const _Float16* p = src + (size_t)(rbase + r) * ld + cbase + 8 * hh;
  v16h f;
#pragma unroll
  for (int i = 0; i < 8; ++i) f[i] = p[i];          // K = 8hh + 0..7
#pragma unroll
  for (int i = 0; i < 8; ++i) f[8 + i] = p[16 + i]; // K = 16 + 8hh + 0..7
  return f;
}

// Same as frag_ld_rm but only K=0..15 valid (head-dim 16 padded to K=32).
__device__ __forceinline__ v16h frag_ld_rm_k16(const _Float16* __restrict__ src,
                                               int ld, int rbase, int cbase) {
  const int lane = threadIdx.x & 31;
  const int r = lane & 15, hh = lane >> 4;
  const _Float16* p = src + (size_t)(rbase + r) * ld + cbase + 8 * hh;
  v16h f;
#pragma unroll
  for (int i = 0; i < 8; ++i) f[i] = p[i];               // K < 16, valid
#pragma unroll
  for (int i = 0; i < 8; ++i) f[8 + i] = (_Float16)0.f;  // K >= 16 -> zero
  return f;
}

// B fragment from a row-major KxN source (e.g. V matrix): lane = column N.
__device__ __forceinline__ v16h frag_ld_b_kn(const _Float16* __restrict__ src,
                                             int ld, int kbase, int nbase) {
  const int lane = threadIdx.x & 31;
  const int n = lane & 15, hh = lane >> 4;
  v16h f;
#pragma unroll
  for (int v = 0; v < 8; ++v) {
    const int k = kbase + ((v < 4) ? 2 * v : 16 + 2 * (v - 4)) + 8 * hh;
    f[2 * v]     = src[(size_t)k * ld + nbase + n];
    f[2 * v + 1] = src[(size_t)(k + 1) * ld + nbase + n];
  }
  return f;
}

// C/D 16x16 f32: lane n=lane&15 (column), VGPR v -> row v + 8*(lane>>4).
__device__ __forceinline__ void frag_st_f16(_Float16* __restrict__ dst, int ld,
                                            int rbase, int cbase, v8f d) {
  const int lane = threadIdx.x & 31;
  const int n = lane & 15, hh = lane >> 4;
#pragma unroll
  for (int v = 0; v < 8; ++v)
    dst[(size_t)(rbase + v + 8 * hh) * ld + cbase + n] = (_Float16)d[v];
}

// ---- prep kernels -----------------------------------------------------------
__global__ void k_f32_to_f16(const float* __restrict__ src,
                             _Float16* __restrict__ dst, int n) {
  const int i = blockIdx.x * 256 + threadIdx.x;
  if (i < n) dst[i] = (_Float16)src[i];
}

// Fused layer-1 weights: wt[n][tap][c] f16, n: [0,64)=ci [64,128)=co
// [128,192)=cg [192,256)=px [256,320)=py.  Sources are fp32 OIHW.
__global__ void k_build_wt(const float* __restrict__ cw1,
                           const float* __restrict__ pxw1,
                           const float* __restrict__ pyw1,
                           _Float16* __restrict__ wt) {
  const int i = blockIdx.x * 256 + threadIdx.x;
  if (i >= kNGate * 576) return;
  const int n = i / 576, rr = i % 576, tap = rr / 64, c = rr % 64;
  const int g = n / 64, cc = n % 64;
  float v;
  if (g == 0)      v = cw1[((size_t)(cc)*128 + c) * 9 + tap];          // ci
  else if (g == 1) v = cw1[((size_t)(128 + cc) * 128 + c) * 9 + tap];  // co
  else if (g == 2) v = cw1[((size_t)(192 + cc) * 128 + c) * 9 + tap];  // cg
  else if (g == 3) v = pxw1[((size_t)cc * 64 + c) * 9 + tap];          // px
  else             v = pyw1[((size_t)cc * 64 + c) * 9 + tap];          // py
  wt[((size_t)n * 9 + tap) * 64 + c] = (_Float16)v;
}

// ---- layer 0: cin==1, h==c==0 -> direct conv + fused gates ------------------
__global__ __launch_bounds__(256) void k_layer0(
    const float* __restrict__ x, const float* __restrict__ cw0,
    const float* __restrict__ cb0, const float* __restrict__ pxw,
    const float* __restrict__ pyw, _Float16* __restrict__ hpre) {
  __shared__ float wci[576], wco[576], wcg[576], wpx[576], wpy[576], bb[192];
  const int tid = threadIdx.x;
  for (int idx = tid; idx < 576; idx += 256) {
    const int c = idx / 9, t = idx % 9;
    wci[idx] = cw0[(size_t)c * 585 + t];            // conv_w0[c, 0, ky, kx]
    wco[idx] = cw0[(size_t)(c + 128) * 585 + t];
    wcg[idx] = cw0[(size_t)(c + 192) * 585 + t];
    wpx[idx] = pxw[idx];
    wpy[idx] = pyw[idx];
  }
  for (int idx = tid; idx < 192; idx += 256) {
    const int g = idx >> 6, c = idx & 63;
    bb[idx] = cb0[(g == 0 ? 0 : (g == 1 ? 128 : 192)) + c];
  }
  __syncthreads();

  const int p = blockIdx.x * 256 + tid;
  const int b = p >> 14, rem = p & 16383, y = rem >> 7, xc = rem & 127;
  float win9[9];
#pragma unroll
  for (int t = 0; t < 9; ++t) {
    const int yy = y + t / 3 - 1, xx = xc + t % 3 - 1;
    win9[t] = (yy >= 0 && yy < kH && xx >= 0 && xx < kW)
                  ? x[(size_t)(b * kH + yy) * kW + xx] : 0.f;
  }
  for (int c = 0; c < 64; ++c) {
    float ci = 0.f, co = 0.f, cg = 0.f, px = 0.f, py = 0.f;
#pragma unroll
    for (int t = 0; t < 9; ++t) {
      const float xv = win9[t];
      ci += wci[c * 9 + t] * xv; co += wco[c * 9 + t] * xv;
      cg += wcg[c * 9 + t] * xv; px += wpx[c * 9 + t] * xv;
      py += wpy[c * 9 + t] * xv;
    }
    const float ig = sigmoidf(ci + bb[c] + px);
    const float og = sigmoidf(co + bb[64 + c] + py);
    const float gg = tanhf(cg + bb[128 + c] + py);
    hpre[(size_t)p * 64 + c] = (_Float16)(og * tanhf(ig * gg));
  }
}

// ---- window attention (WS=8, T=64, C=64, NH=4, HD=16), one WG per window ----
__global__ __launch_bounds__(128) void k_win_attn(
    const _Float16* __restrict__ hin, const _Float16* __restrict__ wqkv,
    const _Float16* __restrict__ wproj, const float* __restrict__ pbias,
    _Float16* __restrict__ hout) {
  __shared__ _Float16 lds_qkv[64 * 192];    // QKV; reused as O[64][64] later
  __shared__ _Float16 lds_ps[4 * 64 * 64];  // per-head P; head0 doubles as X
  _Float16* xs = lds_ps;
  const int tid = threadIdx.x, wave = tid >> 5, lane = tid & 31;
  const int n = lane & 15, hh = lane >> 4;
  const int win = blockIdx.x;
  const int b = win >> 8, wr = (win >> 4) & 15, wc = win & 15;
  const int y0 = wr * 8, x0 = wc * 8;

  { // gather window X [64 tokens][64 ch] via async global->LDS copies
    const int t = tid >> 1, hc = (tid & 1) * 32;
    const int yy = y0 + (t >> 3), xx = x0 + (t & 7);
    const _Float16* src = hin + ((size_t)((b * kH + yy) * kW + xx)) * 64 + hc;
    _Float16* dst = xs + t * 64 + hc;
#pragma unroll
    for (int i = 0; i < 4; ++i) cp16_g2l(src + 8 * i, dst + 8 * i);
    cp_g2l_wait();
  }
  __syncthreads();

  // QKV = X @ Wqkv^T -> [64][192] f16 (wave w owns col tiles 3w..3w+2)
  for (int ct = wave * 3; ct < wave * 3 + 3; ++ct) {
#pragma unroll
    for (int rt = 0; rt < 4; ++rt) {
      v8f acc = {};
#pragma unroll
      for (int ks = 0; ks < 2; ++ks) {
        v16h a = frag_ld_rm(xs, 64, rt * 16, ks * 32);
        v16h bf = frag_ld_rm(wqkv, 64, ct * 16, ks * 32);
        acc = wmma32(a, bf, acc);
      }
      frag_st_f16(lds_qkv, 192, rt * 16, ct * 16, acc);
    }
  }
  __syncthreads();

  // S = Q K^T per head (head = wave); HD=16 zero-padded to K=32
  const int head = wave;
  v8f s[4][4];
#pragma unroll
  for (int rt = 0; rt < 4; ++rt)
#pragma unroll
    for (int ct = 0; ct < 4; ++ct) {
      v8f z = {};
      v16h qa = frag_ld_rm_k16(lds_qkv, 192, rt * 16, head * 16);
      v16h kb = frag_ld_rm_k16(lds_qkv, 192, ct * 16, 64 + head * 16);
      s[rt][ct] = wmma32(qa, kb, z);
    }

  // softmax(S * 0.25) row-wise; rows reduced over 16-lane halves via shfl_xor
  _Float16* ps = lds_ps + head * 64 * 64;
#pragma unroll
  for (int rt = 0; rt < 4; ++rt) {
    float rmax[8], rsum[8];
#pragma unroll
    for (int v = 0; v < 8; ++v) {
      float mv = s[rt][0][v];
#pragma unroll
      for (int ct = 1; ct < 4; ++ct) mv = fmaxf(mv, s[rt][ct][v]);
#pragma unroll
      for (int off = 8; off >= 1; off >>= 1)
        mv = fmaxf(mv, __shfl_xor(mv, off, 32));
      rmax[v] = mv;
    }
#pragma unroll
    for (int ct = 0; ct < 4; ++ct)
#pragma unroll
      for (int v = 0; v < 8; ++v)
        s[rt][ct][v] = exp2f((s[rt][ct][v] - rmax[v]) * 0.36067376022224085f);
#pragma unroll
    for (int v = 0; v < 8; ++v) {
      float sv = 0.f;
#pragma unroll
      for (int ct = 0; ct < 4; ++ct) sv += s[rt][ct][v];
#pragma unroll
      for (int off = 8; off >= 1; off >>= 1) sv += __shfl_xor(sv, off, 32);
      rsum[v] = sv;
    }
#pragma unroll
    for (int ct = 0; ct < 4; ++ct)
#pragma unroll
      for (int v = 0; v < 8; ++v)
        ps[(rt * 16 + v + 8 * hh) * 64 + ct * 16 + n] =
            (_Float16)(s[rt][ct][v] / rsum[v]);
  }

  // O = P @ V  ([64][64] @ [64][16] per head)
  v8f oacc[4];
#pragma unroll
  for (int rt = 0; rt < 4; ++rt) {
    v8f acc = {};
#pragma unroll
    for (int ks = 0; ks < 2; ++ks) {
      v16h a = frag_ld_rm(ps, 64, rt * 16, ks * 32);
      v16h vb = frag_ld_b_kn(lds_qkv + 128 + head * 16, 192, ks * 32, 0);
      acc = wmma32(a, vb, acc);
    }
    oacc[rt] = acc;
  }
  __syncthreads();  // everyone done reading Q/K/V -> reuse lds_qkv for O
#pragma unroll
  for (int rt = 0; rt < 4; ++rt)
    frag_st_f16(lds_qkv, 64, rt * 16, head * 16, oacc[rt]);
  __syncthreads();

  // Y = O @ Wproj^T + bias -> scatter to global (wave = row tile)
#pragma unroll
  for (int ct = 0; ct < 4; ++ct) {
    v8f acc = {};
#pragma unroll
    for (int ks = 0; ks < 2; ++ks) {
      v16h a = frag_ld_rm(lds_qkv, 64, wave * 16, ks * 32);
      v16h bf = frag_ld_rm(wproj, 64, ct * 16, ks * 32);
      acc = wmma32(a, bf, acc);
    }
    const float bv = pbias[ct * 16 + n];
#pragma unroll
    for (int v = 0; v < 8; ++v) {
      const int t = wave * 16 + v + 8 * hh;
      const int yy = y0 + (t >> 3), xx = x0 + (t & 7);
      hout[((size_t)((b * kH + yy) * kW + xx)) * 64 + ct * 16 + n] =
          (_Float16)(acc[v] + bv);
    }
  }
}

// ---- layer 1 fused conv: implicit GEMM [64 pix] x [K=576] x [N=320] ---------
// Weight chunks are staged global->LDS asynchronously (double buffered, one
// copy per workgroup instead of one per wave); A rows stream from global with
// prefetch. LDS is re-used for the fused gate epilogue.
__global__ __launch_bounds__(256) void k_conv_gemm(
    const _Float16* __restrict__ h0, const _Float16* __restrict__ wt,
    const float* __restrict__ cb, _Float16* __restrict__ hpre) {
  // union: B double-buffer 2*[320][kBS] (51200 B) / epilogue [64][320] (40960 B)
  __shared__ __align__(16) _Float16 smem[2 * kNGate * kBS];
  _Float16* lds_d = smem;
  const int tid = threadIdx.x, wave = tid >> 5, lane = tid & 31;
  const int wm = wave & 3, wn = wave >> 2;   // 4 pixel tiles x 2 channel halves
  const int m = lane & 15, hh = lane >> 4;
  const int p = blockIdx.x * 64 + wm * 16 + m;
  const int b = p >> 14, rem = p & 16383, py = rem >> 7, px = rem & 127;

  // stage one 320x32 f16 weight chunk into LDS buffer `buf`
  auto stageB = [&](int kc, int buf) {
    const int tap = kc >> 1, half = (kc & 1) * 32;
    for (int e = tid; e < kNGate * 4; e += 256) {   // 1280 x 16B copies
      const int nrow = e >> 2, q = e & 3;
      cp16_g2l(wt + ((size_t)(nrow * 9 + tap)) * 64 + half + q * 8,
               smem + buf * (kNGate * kBS) + nrow * kBS + q * 8);
    }
  };
  stageB(0, 0);
  cp_g2l_wait();
  __syncthreads();

  v8f acc[10] = {};
  for (int kc = 0; kc < 18; ++kc) {   // K chunks of 32: tap*2 + channel-half
    if (kc + 1 < 18) stageB(kc + 1, (kc + 1) & 1);
    const _Float16* Bs = smem + (kc & 1) * (kNGate * kBS);
    const int tap = kc >> 1;
    const int yy = py + tap / 3 - 1, xx = px + tap % 3 - 1;
    v16h a;
    if (yy >= 0 && yy < kH && xx >= 0 && xx < kW) {
      const _Float16* ap =
          h0 + ((size_t)((b * kH + yy) * kW + xx)) * 64 + (kc & 1) * 32 + 8 * hh;
      const v8h lo = *(const v8h*)ap;
      const v8h hi = *(const v8h*)(ap + 16);
#pragma unroll
      for (int i = 0; i < 8; ++i) { a[i] = lo[i]; a[8 + i] = hi[i]; }
    } else {
#pragma unroll
      for (int i = 0; i < 16; ++i) a[i] = (_Float16)0.f;
    }
    if (kc + 2 < 18) {  // prefetch next A row toward the WGP
      const int tap2 = (kc + 2) >> 1;
      const int yy2 = py + tap2 / 3 - 1, xx2 = px + tap2 % 3 - 1;
      if (yy2 >= 0 && yy2 < kH && xx2 >= 0 && xx2 < kW)
        __builtin_prefetch(h0 + ((size_t)((b * kH + yy2) * kW + xx2)) * 64 +
                               ((kc + 2) & 1) * 32 + 8 * hh, 0, 1);
    }
#pragma unroll
    for (int nt = 0; nt < 10; ++nt) {
      const int ngl = wn * 160 + nt * 16 + m;  // lane's output column
      const _Float16* bp = Bs + ngl * kBS + 8 * hh;
      const v8h blo = *(const v8h*)bp;
      const v8h bhi = *(const v8h*)(bp + 16);
      v16h bf;
#pragma unroll
      for (int i = 0; i < 8; ++i) { bf[i] = blo[i]; bf[8 + i] = bhi[i]; }
      acc[nt] = wmma32(a, bf, acc[nt]);
    }
    cp_g2l_wait();     // this wave's async copies landed
    __syncthreads();   // all waves done reading current buf + copies visible
  }

#pragma unroll
  for (int nt = 0; nt < 10; ++nt)
#pragma unroll
    for (int v = 0; v < 8; ++v)
      lds_d[(wm * 16 + v + 8 * hh) * 320 + wn * 160 + nt * 16 + m] =
          (_Float16)acc[nt][v];
  __syncthreads();

  // gate epilogue: thread -> (pixel t, 16-channel slice)
  const int t = tid >> 2, c16 = (tid & 3) * 16;
  const int pg = blockIdx.x * 64 + t;
#pragma unroll
  for (int c0 = 0; c0 < 16; ++c0) {
    const int c = c16 + c0;
    const float ci  = (float)lds_d[t * 320 + c];
    const float co  = (float)lds_d[t * 320 + 64 + c];
    const float cg  = (float)lds_d[t * 320 + 128 + c];
    const float pxv = (float)lds_d[t * 320 + 192 + c];
    const float pyv = (float)lds_d[t * 320 + 256 + c];
    const float ig = sigmoidf(ci + pxv + cb[c]);
    const float og = sigmoidf(co + pyv + cb[128 + c]);
    const float gg = tanhf(cg + pyv + cb[192 + c]);
    hpre[(size_t)pg * 64 + c] = (_Float16)(og * tanhf(ig * gg));
  }
}

// ---- final 1x1 conv ---------------------------------------------------------
__global__ __launch_bounds__(256) void k_out(const _Float16* __restrict__ h1,
                                             const float* __restrict__ ow,
                                             const float* __restrict__ ob,
                                             float* __restrict__ out) {
  const int p = blockIdx.x * 256 + threadIdx.x;
  float acc = ob[0];
#pragma unroll
  for (int c = 0; c < 64; ++c) acc += (float)h1[(size_t)p * 64 + c] * ow[c];
  out[p] = acc;
}

extern "C" void kernel_launch(void* const* d_in, const int* in_sizes, int n_in,
                              void* d_out, int out_size, void* d_ws,
                              size_t ws_size, hipStream_t stream) {
  (void)in_sizes; (void)n_in; (void)out_size; (void)ws_size;
  const float* x       = (const float*)d_in[0];
  const float* conv_w0 = (const float*)d_in[1];
  const float* conv_b0 = (const float*)d_in[2];
  const float* px_w0   = (const float*)d_in[3];
  const float* py_w0   = (const float*)d_in[4];
  const float* qkv_w0  = (const float*)d_in[5];
  const float* proj_w0 = (const float*)d_in[6];
  const float* proj_b0 = (const float*)d_in[7];
  const float* conv_w1 = (const float*)d_in[8];
  const float* conv_b1 = (const float*)d_in[9];
  const float* px_w1   = (const float*)d_in[10];
  const float* py_w1   = (const float*)d_in[11];
  const float* qkv_w1  = (const float*)d_in[12];
  const float* proj_w1 = (const float*)d_in[13];
  const float* proj_b1 = (const float*)d_in[14];
  const float* out_w   = (const float*)d_in[15];
  const float* out_b   = (const float*)d_in[16];
  float* out = (float*)d_out;

  char* ws = (char*)d_ws;
  _Float16* bufA = (_Float16*)ws;                            // 32 MB h_pre
  _Float16* bufB = (_Float16*)(ws + (size_t)kPix * 64 * 2);  // 32 MB h
  _Float16* wt   = (_Float16*)(ws + (size_t)kPix * 64 * 4);  // fused conv1 W
  _Float16* wq0  = wt + (size_t)kNGate * 576;
  _Float16* wp0  = wq0 + 192 * 64;
  _Float16* wq1  = wp0 + 64 * 64;
  _Float16* wp1  = wq1 + 192 * 64;

  k_f32_to_f16<<<48, 256, 0, stream>>>(qkv_w0, wq0, 192 * 64);
  k_f32_to_f16<<<16, 256, 0, stream>>>(proj_w0, wp0, 64 * 64);
  k_f32_to_f16<<<48, 256, 0, stream>>>(qkv_w1, wq1, 192 * 64);
  k_f32_to_f16<<<16, 256, 0, stream>>>(proj_w1, wp1, 64 * 64);
  k_build_wt<<<(kNGate * 576 + 255) / 256, 256, 0, stream>>>(conv_w1, px_w1,
                                                             py_w1, wt);
  k_layer0<<<kPix / 256, 256, 0, stream>>>(x, conv_w0, conv_b0, px_w0, py_w0,
                                           bufA);
  k_win_attn<<<kPix / 64, 128, 0, stream>>>(bufA, wq0, wp0, proj_b0, bufB);
  k_conv_gemm<<<kPix / 64, 256, 0, stream>>>(bufB, wt, conv_b1, bufA);
  k_win_attn<<<kPix / 64, 128, 0, stream>>>(bufA, wq1, wp1, proj_b1, bufB);
  k_out<<<kPix / 256, 256, 0, stream>>>(bufB, out_w, out_b, out);
}
